// DocREModel_24790551232533
// MI455X (gfx1250) — matrix operand order repaired and checked by reading, not compiled, not run on hardware
//
#include <hip/hip_runtime.h>
#include <hip/hip_bf16.h>
#include <math.h>

typedef _Float16 f16;
typedef __attribute__((ext_vector_type(8)))  _Float16 v8h;
typedef __attribute__((ext_vector_type(16))) _Float16 v16h;
typedef __attribute__((ext_vector_type(8)))  float    v8f;

#define BB 4
#define HH 12
#define CC 1024
#define DD 768
#define EE 30
#define MMENT 8
#define PP 600
#define NLAB 97
#define NREL 96
#define EMBD 768
#define BLKD 64
#define NPAIRS 2400      // BB*PP
#define KCAT 1536        // 2*DD
#define KBIL 49152       // EMBD*BLKD

// ---------------------------------------------------------------------------
// Fragment helpers (wave32 WMMA f16 16x16x32). Per CDNA5 ISA 7.12.2 each
// lane's 16 A-halves are two contiguous 8-half groups -> two b128 loads.
// B is consumed from an N-major (transposed) copy so its 16 halves per lane
// are two contiguous 8-half groups (K+0, K+16), as for A -> two b128 loads.
// ---------------------------------------------------------------------------
static __device__ __forceinline__ v16h frag_from2(v8h lo, v8h hi) {
  return __builtin_shufflevector(lo, hi, 0, 1, 2, 3, 4, 5, 6, 7,
                                 8, 9, 10, 11, 12, 13, 14, 15);
}

static __device__ __forceinline__ v16h load_frag_a(const f16* __restrict__ Arow,
                                                   int k0, int hs) {
  const f16* p = Arow + k0 + 8 * hs;         // 16B aligned
  v8h lo = *(const v8h*)p;                   // K = k0+8hs .. +7
  v8h hi = *(const v8h*)(p + 16);            // K = k0+16+8hs .. +7
  return frag_from2(lo, hi);
}

static __device__ __forceinline__ v16h load_frag_bT(const f16* __restrict__ Bn,
                                                    int k0, int hs) {
  const f16* p = Bn + k0 + 8 * hs;           // two 8-half groups (K+0, K+16), 16B aligned
  v8h lo = *(const v8h*)p;
  v8h hi = *(const v8h*)(p + 16);
  return frag_from2(lo, hi);
}

// ---------------------------------------------------------------------------
// Generic WMMA GEMM vs transposed B: D = epilogue(A[MxK] * BT[NxK]^T).
// One 16x64 macro-tile per wave (1 A frag -> 4 WMMA per K-step).
// mode 0: out16 = (f16)acc              (rs)
// mode 1: out16 = (f16)tanh(acc + bias) (hz/tz)
// ---------------------------------------------------------------------------
__global__ void wmma_gemm_nt(const f16* __restrict__ A, const f16* __restrict__ BT,
                             const float* __restrict__ bias, f16* __restrict__ out16,
                             int M, int N, int K, int lda, int ldbt, int ldo, int mode) {
  const int wave = blockIdx.x * (blockDim.x >> 5) + (threadIdx.x >> 5);
  const int mt = (M + 15) >> 4;
  const int nt = (N + 15) >> 4;
  const int ng = (nt + 3) >> 2;
  if (wave >= mt * ng) return;               // wave-uniform: EXEC stays all ones
  const int row0 = (wave / ng) * 16;
  const int colg = (wave % ng) * 64;

  const int lane = threadIdx.x & 31;
  const int hs   = lane >> 4;
  int m = row0 + (lane & 15);
  if (m >= M) m = M - 1;                     // clamp: masked rows never stored
  const f16* Arow = A + (size_t)m * lda;

  const f16* Bp[4];
#pragma unroll
  for (int j = 0; j < 4; ++j) {
    int n = colg + j * 16 + (lane & 15);
    if (n >= N) n = N - 1;                   // clamp: masked cols never stored
    Bp[j] = BT + (size_t)n * ldbt;
  }

  v8f acc[4] = {{}, {}, {}, {}};
  for (int k0 = 0; k0 < K; k0 += 32) {
    __builtin_prefetch(Arow + k0 + 64, 0, 0);          // global_prefetch_b8
    v16h a = load_frag_a(Arow, k0, hs);
#pragma unroll
    for (int j = 0; j < 4; ++j) {
      v16h b = load_frag_bT(Bp[j], k0, hs);
      acc[j] = __builtin_amdgcn_wmma_f32_16x16x32_f16(false, a, false, b,
                                                      (short)0, acc[j], false, false);
    }
  }

#pragma unroll
  for (int j = 0; j < 4; ++j) {
    const int col = colg + j * 16 + (lane & 15);
    if (col < N) {
#pragma unroll
      for (int r = 0; r < 8; ++r) {
        const int row = row0 + hs * 8 + r;
        if (row < M) {
          float v = acc[j][r];
          if (mode == 1) v = tanhf(v + bias[col]);
          out16[(size_t)row * ldo + col] = (f16)v;
        }
      }
    }
  }
}

// ---------------------------------------------------------------------------
// Fused block-bilinear logits GEMM. A-fragments synthesized on the fly:
// bl[n, blk*4096 + i*64 + j] = hz[n, blk*64+i] * tz[n, blk*64+j].
// Within a lane's 8-half group, (blk, i) are constant and j is contiguous:
// 1 scalar hz load + 1 b128 tz load + packed f16 multiply per group.
// ---------------------------------------------------------------------------
__global__ void wmma_gemm_bilinear(const f16* __restrict__ hz, const f16* __restrict__ tz,
                                   const f16* __restrict__ BT, const float* __restrict__ bil_b,
                                   float* __restrict__ logits) {
  const int wave = blockIdx.x * (blockDim.x >> 5) + (threadIdx.x >> 5);
  const int mt = NPAIRS / 16;                // 150
  const int nt = (NLAB + 15) >> 4;           // 7
  const int ng = (nt + 3) >> 2;              // 2
  if (wave >= mt * ng) return;
  const int row0 = (wave / ng) * 16;
  const int colg = (wave % ng) * 64;

  const int lane = threadIdx.x & 31;
  const int hs   = lane >> 4;
  const int m    = row0 + (lane & 15);       // NPAIRS % 16 == 0
  const f16* hrow = hz + (size_t)m * EMBD;
  const f16* trow = tz + (size_t)m * EMBD;

  const f16* Bp[4];
#pragma unroll
  for (int j = 0; j < 4; ++j) {
    int n = colg + j * 16 + (lane & 15);
    if (n >= NLAB) n = NLAB - 1;
    Bp[j] = BT + (size_t)n * KBIL;
  }

  v8f acc[4] = {{}, {}, {}, {}};
  for (int k0 = 0; k0 < KBIL; k0 += 32) {
    const int kkA = k0 + 8 * hs;             // group 0: e = 0..7
    const int kkB = kkA + 16;                // group 1: e = 8..15
    const int blkA = kkA >> 12, iA = (kkA >> 6) & 63, jA = kkA & 63;
    const int blkB = kkB >> 12, iB = (kkB >> 6) & 63, jB = kkB & 63;
    const f16 hA = hrow[blkA * BLKD + iA];
    const f16 hB = hrow[blkB * BLKD + iB];
    v8h tA = *(const v8h*)(trow + blkA * BLKD + jA);   // jA multiple of 8
    v8h tB = *(const v8h*)(trow + blkB * BLKD + jB);
    v8h hAv = {hA, hA, hA, hA, hA, hA, hA, hA};
    v8h hBv = {hB, hB, hB, hB, hB, hB, hB, hB};
    v16h a = frag_from2(hAv * tA, hBv * tB);
#pragma unroll
    for (int j = 0; j < 4; ++j) {
      v16h b = load_frag_bT(Bp[j], k0, hs);
      acc[j] = __builtin_amdgcn_wmma_f32_16x16x32_f16(false, a, false, b,
                                                      (short)0, acc[j], false, false);
    }
  }

#pragma unroll
  for (int j = 0; j < 4; ++j) {
    const int col = colg + j * 16 + (lane & 15);
    if (col < NLAB) {
#pragma unroll
      for (int r = 0; r < 8; ++r) {
        const int row = row0 + hs * 8 + r;
        logits[(size_t)row * NLAB + col] = acc[j][r] + bil_b[col];
      }
    }
  }
}

// ---------------------------------------------------------------------------
// Conversion / transpose kernels (one-time, L2-resident afterwards)
// ---------------------------------------------------------------------------
__global__ void f32_to_f16_kernel(const float* __restrict__ in, f16* __restrict__ out, int n) {
  int t = blockIdx.x * blockDim.x + threadIdx.x;
  if (t < n) out[t] = (f16)in[t];
}

// seqT[b][d][c] = seq[b][c][d]
__global__ void seq_transpose_kernel(const float* __restrict__ in, f16* __restrict__ out) {
  int t = blockIdx.x * blockDim.x + threadIdx.x;
  if (t >= BB * DD * CC) return;
  const int c = t % CC;
  const int rem = t / CC;
  const int d = rem % DD;
  const int b = rem / DD;
  out[t] = (f16)in[((size_t)b * CC + c) * DD + d];
}

// out[c][r] = in[r][c]; in is [R x Ccols] row-major
__global__ void w_transpose_kernel(const float* __restrict__ in, f16* __restrict__ out,
                                   int R, int Ccols) {
  int t = blockIdx.x * blockDim.x + threadIdx.x;
  if (t >= R * Ccols) return;
  const int r = t % R;
  const int c = t / R;
  out[t] = (f16)in[(size_t)r * Ccols + c];
}

// ---------------------------------------------------------------------------
// Pooling / gather kernels
// ---------------------------------------------------------------------------
__global__ void ent_emb_kernel(const float* __restrict__ seq, const int* __restrict__ mi,
                               const float* __restrict__ mk, float* __restrict__ out) {
  int t = blockIdx.x * blockDim.x + threadIdx.x;
  if (t >= BB * EE * DD) return;
  const int d  = t % DD;
  const int be = t / DD;
  const int b  = be / EE;
  const int*   mip = mi + be * MMENT;
  const float* mkp = mk + be * MMENT;
  float v[MMENT];
#pragma unroll
  for (int m = 0; m < MMENT; ++m) {
    const int idx = mip[m] + 1;                       // OFFSET
    v[m] = seq[((size_t)b * CC + idx) * DD + d];
  }
  float mx = -3.402823e38f;
#pragma unroll
  for (int m = 0; m < MMENT; ++m)
    mx = fmaxf(mx, (mkp[m] > 0.0f) ? v[m] : -3.402823e38f);
  float s = 0.0f;
#pragma unroll
  for (int m = 0; m < MMENT; ++m)
    s += (mkp[m] > 0.0f) ? expf(v[m] - mx) : 0.0f;
  out[t] = mx + logf(s);
}

__global__ void ent_att_kernel(const float* __restrict__ att, const int* __restrict__ mi,
                               const float* __restrict__ mk, float* __restrict__ out) {
  int t = blockIdx.x * blockDim.x + threadIdx.x;
  if (t >= BB * EE * HH * CC) return;
  const int c  = t % CC;
  int r        = t / CC;
  const int h  = r % HH;
  const int be = r / HH;
  const int b  = be / EE;
  const int*   mip = mi + be * MMENT;
  const float* mkp = mk + be * MMENT;
  float s = 0.0f, cnt = 0.0f;
#pragma unroll
  for (int m = 0; m < MMENT; ++m) {
    const float w = mkp[m];
    const int idx = mip[m] + 1;
    cnt += w;
    s += w * att[(((size_t)b * HH + h) * CC + idx) * CC + c];
  }
  out[t] = s / cnt;
}

// ht_att[b,p,c] (mean over heads, sum-normalized) -> f16. One block per pair.
__global__ void htatt_kernel(const float* __restrict__ entatt, const int* __restrict__ hts,
                             f16* __restrict__ out) {
  const int bp = blockIdx.x;                 // b*PP + p
  const int b  = bp / PP;
  const int hi = hts[bp * 2 + 0];
  const int ti = hts[bp * 2 + 1];
  const float* ha = entatt + ((size_t)(b * EE + hi)) * HH * CC;
  const float* ta = entatt + ((size_t)(b * EE + ti)) * HH * CC;

  float vals[CC / 256];
  float part = 0.0f;
#pragma unroll
  for (int q = 0; q < CC / 256; ++q) {
    const int c = threadIdx.x + q * 256;
    float s = 0.0f;
#pragma unroll
    for (int h = 0; h < HH; ++h) s += ha[h * CC + c] * ta[h * CC + c];
    s *= (1.0f / HH);
    vals[q] = s;
    part += s;
  }
  __shared__ float red[256];
  red[threadIdx.x] = part;
  __syncthreads();
  for (int w = 128; w > 0; w >>= 1) {
    if (threadIdx.x < w) red[threadIdx.x] += red[threadIdx.x + w];
    __syncthreads();
  }
  const float inv = 1.0f / (red[0] + 1e-5f);
#pragma unroll
  for (int q = 0; q < CC / 256; ++q)
    out[(size_t)bp * CC + threadIdx.x + q * 256] = (f16)(vals[q] * inv);
}

// cat_h = [hs | rs], cat_t = [ts | rs] in f16
__global__ void build_cat_kernel(const float* __restrict__ entemb, const f16* __restrict__ rs16,
                                 const int* __restrict__ hts,
                                 f16* __restrict__ cat_h, f16* __restrict__ cat_t) {
  int t = blockIdx.x * blockDim.x + threadIdx.x;
  if (t >= NPAIRS * KCAT) return;
  const int n   = t / KCAT;
  const int col = t % KCAT;
  const int b   = n / PP;
  if (col < DD) {
    const int hi = hts[n * 2 + 0];
    const int ti = hts[n * 2 + 1];
    cat_h[t] = (f16)entemb[((size_t)(b * EE + hi)) * DD + col];
    cat_t[t] = (f16)entemb[((size_t)(b * EE + ti)) * DD + col];
  } else {
    const f16 v = rs16[(size_t)n * DD + (col - DD)];
    cat_h[t] = v;
    cat_t[t] = v;
  }
}

__global__ void zero_scalar_kernel(float* p) { *p = 0.0f; }

// Per-class S-PU risk; one block per relation class, atomicAdd into out[0].
__global__ void pu_risk_kernel(const float* __restrict__ logits, const int* __restrict__ labels,
                               const float* __restrict__ pl, const float* __restrict__ po,
                               float* __restrict__ out) {
  const int r = blockIdx.x;                  // 0..NREL-1
  float sn = 0.0f, spp = 0.0f, spn = 0.0f;
  int cp = 0;
  for (int n = threadIdx.x; n < NPAIRS; n += blockDim.x) {
    const float s  = logits[(size_t)n * NLAB + r + 1] - logits[(size_t)n * NLAB];
    const bool pos = labels[(size_t)n * NLAB + r + 1] == 1;
    const float lp = 0.25f * (s - 1.0f) * (s - 1.0f);       // sign=+1, MARGIN=1
    const float ln = 0.25f * (-s - 1.0f) * (-s - 1.0f);     // sign=-1
    if (pos) { spp += lp; spn += ln; cp++; } else { sn += ln; }
  }
  __shared__ float r0[256], r1[256], r2[256];
  __shared__ int   r3[256];
  r0[threadIdx.x] = sn; r1[threadIdx.x] = spp; r2[threadIdx.x] = spn; r3[threadIdx.x] = cp;
  __syncthreads();
  for (int w = 128; w > 0; w >>= 1) {
    if (threadIdx.x < w) {
      r0[threadIdx.x] += r0[threadIdx.x + w];
      r1[threadIdx.x] += r1[threadIdx.x + w];
      r2[threadIdx.x] += r2[threadIdx.x + w];
      r3[threadIdx.x] += r3[threadIdx.x + w];
    }
    __syncthreads();
  }
  if (threadIdx.x == 0) {
    const int   cpos = r3[0];
    const int   cneg = NPAIRS - cpos;
    const float sq_neg   = (cneg > 0) ? r0[0] / (float)cneg : 0.0f;
    const float sq_pos_p = (cpos > 0) ? r1[0] / (float)cpos : 0.0f;
    const float sq_pos_n = (cpos > 0) ? r2[0] / (float)cpos : 0.0f;
    const float poV = po[r], plV = pl[r];
    const float w   = sqrtf((1.0f - poV) / poV);
    const float pu  = (poV - plV) / (1.0f - plV);
    const float risk1 = (1.0f - poV) / (1.0f - pu) * sq_neg
                      - (pu - pu * poV) / (1.0f - pu) * sq_pos_n;
    const float risk2 = poV * sq_pos_p * w;
    const float risk  = (risk1 < 0.0f) ? -risk1 : (risk1 + risk2);  // BETA=0, GAMMA=1
    atomicAdd(out, risk);
  }
}

// ---------------------------------------------------------------------------
// Host orchestration
// ---------------------------------------------------------------------------
extern "C" void kernel_launch(void* const* d_in, const int* in_sizes, int n_in,
                              void* d_out, int out_size, void* d_ws, size_t ws_size,
                              hipStream_t stream) {
  (void)in_sizes; (void)n_in; (void)out_size; (void)ws_size;

  const float* seq      = (const float*)d_in[0];
  const float* att      = (const float*)d_in[1];
  const int*   mi       = (const int*)  d_in[2];
  const float* mk       = (const float*)d_in[3];
  const int*   hts      = (const int*)  d_in[4];
  const int*   labels   = (const int*)  d_in[5];
  const float* priors_l = (const float*)d_in[6];
  const float* priors_o = (const float*)d_in[7];
  const float* head_W   = (const float*)d_in[8];
  const float* head_b   = (const float*)d_in[9];
  const float* tail_W   = (const float*)d_in[10];
  const float* tail_b   = (const float*)d_in[11];
  const float* bil_W    = (const float*)d_in[12];
  const float* bil_b    = (const float*)d_in[13];

  float* out    = (float*)d_out;       // [0]=risk, [1..]=logits (2400x97)
  float* logits = out + 1;

  char* ws = (char*)d_ws;
  size_t off = 0;
  auto carve = [&](size_t bytes) -> void* {
    void* p = ws + off;
    off = (off + bytes + 255) & ~(size_t)255;
    return p;
  };
  f16*   seqT16   = (f16*)  carve((size_t)BB * DD * CC * 2);      // [B][D][C]
  f16*   headWT16 = (f16*)  carve((size_t)EMBD * KCAT * 2);       // [EMB][2D]
  f16*   tailWT16 = (f16*)  carve((size_t)EMBD * KCAT * 2);
  f16*   bilWT16  = (f16*)  carve((size_t)NLAB * KBIL * 2);       // [97][49152]
  float* entemb   = (float*)carve((size_t)BB * EE * DD * 4);
  float* entatt   = (float*)carve((size_t)BB * EE * HH * CC * 4);
  f16*   htatt16  = (f16*)  carve((size_t)NPAIRS * CC * 2);
  f16*   rs16     = (f16*)  carve((size_t)NPAIRS * DD * 2);
  f16*   cat_h    = (f16*)  carve((size_t)NPAIRS * KCAT * 2);
  f16*   cat_t    = (f16*)  carve((size_t)NPAIRS * KCAT * 2);
  f16*   hz16     = (f16*)  carve((size_t)NPAIRS * EMBD * 2);
  f16*   tz16     = (f16*)  carve((size_t)NPAIRS * EMBD * 2);

  auto cdiv = [](int a, int b) { return (a + b - 1) / b; };

  // 1) transposed f16 operand copies
  seq_transpose_kernel<<<cdiv(BB * DD * CC, 256), 256, 0, stream>>>(seq, seqT16);
  w_transpose_kernel<<<cdiv(KCAT * EMBD, 256), 256, 0, stream>>>(head_W, headWT16, KCAT, EMBD);
  w_transpose_kernel<<<cdiv(KCAT * EMBD, 256), 256, 0, stream>>>(tail_W, tailWT16, KCAT, EMBD);
  w_transpose_kernel<<<cdiv(KBIL * NLAB, 256), 256, 0, stream>>>(bil_W, bilWT16, KBIL, NLAB);

  // 2) entity pooling
  ent_emb_kernel<<<cdiv(BB * EE * DD, 256), 256, 0, stream>>>(seq, mi, mk, entemb);
  ent_att_kernel<<<cdiv(BB * EE * HH * CC, 256), 256, 0, stream>>>(att, mi, mk, entatt);

  // 3) pair attention -> normalized ht_att (f16)
  htatt_kernel<<<NPAIRS, 256, 0, stream>>>(entatt, hts, htatt16);

  // 4) rs = ht_att @ seq per batch (WMMA)
  for (int b = 0; b < BB; ++b) {
    const int waves = cdiv(PP, 16) * cdiv(DD / 16, 4);   // 38 * 12
    wmma_gemm_nt<<<cdiv(waves, 4), 128, 0, stream>>>(
        htatt16 + (size_t)b * PP * CC, seqT16 + (size_t)b * DD * CC, nullptr,
        rs16 + (size_t)b * PP * DD, PP, DD, CC, CC, CC, DD, 0);
  }

  // 5) [hs|rs], [ts|rs]
  build_cat_kernel<<<cdiv(NPAIRS * KCAT, 256), 256, 0, stream>>>(entemb, rs16, hts, cat_h, cat_t);

  // 6) hz/tz = tanh(cat @ W + b) (WMMA, fused epilogue)
  {
    const int waves = (NPAIRS / 16) * cdiv(EMBD / 16, 4);  // 150 * 12
    wmma_gemm_nt<<<cdiv(waves, 4), 128, 0, stream>>>(
        cat_h, headWT16, head_b, hz16, NPAIRS, EMBD, KCAT, KCAT, KCAT, EMBD, 1);
    wmma_gemm_nt<<<cdiv(waves, 4), 128, 0, stream>>>(
        cat_t, tailWT16, tail_b, tz16, NPAIRS, EMBD, KCAT, KCAT, KCAT, EMBD, 1);
  }

  // 7) logits = bl(hz,tz) @ bil_W + bil_b, bilinear fused into A fragments
  {
    const int waves = (NPAIRS / 16) * 2;                   // 150 * ceil(7/4)
    wmma_gemm_bilinear<<<cdiv(waves, 4), 128, 0, stream>>>(hz16, tz16, bilWT16, bil_b, logits);
  }

  // 8) PU risk -> out[0]
  zero_scalar_kernel<<<1, 1, 0, stream>>>(out);
  pu_risk_kernel<<<NREL, 256, 0, stream>>>(logits, labels, priors_l, priors_o, out);
}
